// GoodfireSAE_70300024700996
// MI455X (gfx1250) — compile-verified, matrix operand order checked by reading, not run on hardware
//
#include <hip/hip_runtime.h>

// ---------------------------------------------------------------------------
// GoodfireSAE forward for MI455X (gfx1250, wave32).
//   Pass 1: feat  = relu(x @ W_enc^T + b_enc)      [4096 x 32768]
//   Pass 2: per-token top-k mask in place on feat  (bit-pattern binary search)
//   Pass 3: recon = feat @ W_dec^T + b_dec         [4096 x 4096]
// Roofline: ~2-3 GB HBM traffic @ 23.3 TB/s dominates; ~2.2 TFLOP of matmul is
// negligible even at f32-WMMA rates -> use V_WMMA_F32_16X16X4_F32 (exact f32,
// zero conversion VALU) + raw async global->LDS staging (ASYNCcnt).
// Falls back to the codegen-proven bf16 WMMA path if builtins are absent.
// ---------------------------------------------------------------------------

typedef __attribute__((ext_vector_type(2)))  float         v2f;
typedef __attribute__((ext_vector_type(4)))  float         v4f;
typedef __attribute__((ext_vector_type(8)))  float         v8f;
typedef __attribute__((ext_vector_type(16))) __bf16        v16bf;
typedef __attribute__((ext_vector_type(4)))  unsigned int  v4u;
typedef __attribute__((ext_vector_type(4)))  int           v4i;

#if defined(__AMDGCN__) && __has_builtin(__builtin_amdgcn_wmma_f32_16x16x4_f32)
#define USE_F32_WMMA 1
#else
#define USE_F32_WMMA 0
#endif

#if defined(__AMDGCN__) && __has_builtin(__builtin_amdgcn_global_load_async_to_lds_b128)
#define HAVE_ASYNC 1
#else
#define HAVE_ASYNC 0
#endif

#if HAVE_ASYNC
__device__ __forceinline__ void async_cp16(const float* g, float* l) {
  // one 16B lane-chunk: global -> LDS, DMA'd, tracked by ASYNCcnt.
  // builtin signature (from probe diagnostic): param0 = AS1 v4i*, param1 = LDS v4i*
  __builtin_amdgcn_global_load_async_to_lds_b128(
      (__attribute__((address_space(1))) v4i*)g,
      (__attribute__((address_space(3))) v4i*)l, 0, 0);
}
#endif

template <int N>
__device__ __forceinline__ void wait_async() {
#if HAVE_ASYNC
#if __has_builtin(__builtin_amdgcn_s_wait_asynccnt)
  __builtin_amdgcn_s_wait_asynccnt(N);
#else
  asm volatile("s_wait_asynccnt %0" ::"n"(N) : "memory");
#endif
#endif
}

// --- tiling (shared by both device paths so host launch config is fixed) ----
#define BM 128
#define BN 64
#define BK 32
#define FP 36   // f32 LDS row pitch (32 data + 4 pad) = 144B: 16B-aligned, conflict-free
#define HP 40   // bf16 LDS row pitch (u16) = 80B

#if !USE_F32_WMMA
__device__ __forceinline__ unsigned short f2bf(float f) {
  __bf16 b = (__bf16)f;                    // native fptrunc (RNE)
  unsigned short u; __builtin_memcpy(&u, &b, 2); return u;
}
__device__ __forceinline__ v16bf load_frag(const unsigned short* rowp, int h) {
  // ISA 16-bit A/B 16x32 layout: lanes0-15 K[0..8)+K[16..24), lanes16-31 K[8..16)+K[24..32)
  v4u c0 = *(const v4u*)(rowp + h * 8);
  v4u c1 = *(const v4u*)(rowp + 16 + h * 8);
  v16bf r;
  __builtin_memcpy(&r, &c0, 16);
  __builtin_memcpy(reinterpret_cast<char*>(&r) + 16, &c1, 16);
  return r;
}
#endif

// C[m,n] = act( sum_k A[m,k]*B[n,k] + bias[n] ),  A:MxK, B:NxK row-major.
// 256 threads = 8 waves (2x4); wave tile 64x16 -> 4 v8f accumulators.
__global__ __launch_bounds__(256) void gemm_wmma(
    const float* __restrict__ A, const float* __restrict__ B,
    const float* __restrict__ bias, float* __restrict__ C,
    int M, int N, int K, int do_relu)
{
  const int t    = threadIdx.x;
  const int lane = t & 31;
  const int wave = t >> 5;
  const int wm   = wave >> 2;          // 0..1 -> 64-row slab
  const int wn   = wave & 3;           // 0..3 -> 16-col slab
  const int r16  = lane & 15;
  const int h    = lane >> 4;
  const int rowBase = blockIdx.x * BM; // M fastest -> streamed weights L2-shared
  const int colBase = blockIdx.y * BN;
  const int NT = K / BK;

#if USE_F32_WMMA
  // ------------------ exact-f32 WMMA path with async staging ------------------
  __shared__ float lA[2][BM * FP];
  __shared__ float lB[2][BN * FP];

  const int arow = t >> 1, ahalf = t & 1;      // A: 128 rows x 2 half-rows
  const int brow = t >> 2, bq    = t & 3;      // B: 64 rows  x 4 quarter-rows
  const float* gA = A + (size_t)(rowBase + arow) * K + ahalf * 16;
  const float* gB = B + (size_t)(colBase + brow) * K + bq * 8;

  v8f acc[4];
#pragma unroll
  for (int i = 0; i < 4; ++i) acc[i] = (v8f)0.0f;

  auto stage = [&](int kt, int buf) {           // 6 x 16B chunks per thread
    const float* ga = gA + (size_t)kt * BK;
    const float* gb = gB + (size_t)kt * BK;
    float* la = &lA[buf][arow * FP + ahalf * 16];
    float* lb = &lB[buf][brow * FP + bq * 8];
#if HAVE_ASYNC
#pragma unroll
    for (int j = 0; j < 4; ++j) async_cp16(ga + j * 4, la + j * 4);
#pragma unroll
    for (int j = 0; j < 2; ++j) async_cp16(gb + j * 4, lb + j * 4);
#else
    v4f ra[4], rb[2];
#pragma unroll
    for (int j = 0; j < 4; ++j) ra[j] = ((const v4f*)ga)[j];
#pragma unroll
    for (int j = 0; j < 2; ++j) rb[j] = ((const v4f*)gb)[j];
#pragma unroll
    for (int j = 0; j < 4; ++j) ((v4f*)la)[j] = ra[j];
#pragma unroll
    for (int j = 0; j < 2; ++j) ((v4f*)lb)[j] = rb[j];
#endif
  };

  stage(0, 0);
  int buf = 0;
  for (int kt = 0; kt < NT; ++kt) {
    if (kt + 1 < NT) {
      stage(kt + 1, buf ^ 1);
      wait_async<6>();                 // in-order: first 6 (tile kt) have landed
    } else {
      wait_async<0>();
    }
    if (kt + 2 < NT) {
      __builtin_prefetch(gA + (size_t)(kt + 2) * BK, 0, 1);
      __builtin_prefetch(gB + (size_t)(kt + 2) * BK, 0, 1);
    }
    __syncthreads();

    const float* Af = lA[buf];
    const float* Bf = lB[buf];
    // f32 A 16x4 layout: lane m=l&15, k = 2h+{0,1}  -> contiguous v2f per lane
#pragma unroll
    for (int s = 0; s < 8; ++s) {
      v2f bfrag = *(const v2f*)(Bf + (wn * 16 + r16) * FP + s * 4 + h * 2);
#pragma unroll
      for (int i = 0; i < 4; ++i) {
        v2f afrag = *(const v2f*)(Af + (wm * 64 + i * 16 + r16) * FP + s * 4 + h * 2);
        acc[i] = __builtin_amdgcn_wmma_f32_16x16x4_f32(
            false, afrag, false, bfrag, (short)0, acc[i], false, false);
      }
    }
    __syncthreads();
    buf ^= 1;
  }

  const int col = colBase + wn * 16 + r16;
  const float bv = bias[col];
#pragma unroll
  for (int i = 0; i < 4; ++i) {
    const int row0 = rowBase + wm * 64 + i * 16 + h * 8;
#pragma unroll
    for (int p = 0; p < 8; ++p) {
      float v = acc[i][p] + bv;
      if (do_relu) v = fmaxf(v, 0.0f);
      C[(size_t)(row0 + p) * N + col] = v;
    }
  }

#else
  // ------------------ bf16 WMMA fallback (codegen-proven) ------------------
  __shared__ unsigned short lA[BM * HP];
  __shared__ unsigned short lB[BN * HP];

  const int arow = t >> 1, ahalf = t & 1;
  const int brow = t >> 2, bq    = t & 3;
  const float* gA = A + (size_t)(rowBase + arow) * K + ahalf * 16;
  const float* gB = B + (size_t)(colBase + brow) * K + bq * 8;

  v8f acc[4];
#pragma unroll
  for (int i = 0; i < 4; ++i) acc[i] = (v8f)0.0f;

  for (int kt = 0; kt < NT; ++kt) {
    {
      const v4f* pa = (const v4f*)(gA + (size_t)kt * BK);
      unsigned short ha[16] __attribute__((aligned(16)));
#pragma unroll
      for (int j = 0; j < 4; ++j) {
        v4f v = pa[j];
        ha[4*j+0] = f2bf(v.x); ha[4*j+1] = f2bf(v.y);
        ha[4*j+2] = f2bf(v.z); ha[4*j+3] = f2bf(v.w);
      }
      v4u* da = (v4u*)(lA + arow * HP + ahalf * 16);
      da[0] = *(const v4u*)(ha + 0);
      da[1] = *(const v4u*)(ha + 8);

      const v4f* pb = (const v4f*)(gB + (size_t)kt * BK);
      unsigned short hb[8] __attribute__((aligned(16)));
#pragma unroll
      for (int j = 0; j < 2; ++j) {
        v4f v = pb[j];
        hb[4*j+0] = f2bf(v.x); hb[4*j+1] = f2bf(v.y);
        hb[4*j+2] = f2bf(v.z); hb[4*j+3] = f2bf(v.w);
      }
      *(v4u*)(lB + brow * HP + bq * 8) = *(const v4u*)hb;
    }
    __syncthreads();

    v16bf bfr = load_frag(lB + (wn * 16 + r16) * HP, h);
#pragma unroll
    for (int i = 0; i < 4; ++i) {
      v16bf af = load_frag(lA + (wm * 64 + i * 16 + r16) * HP, h);
      acc[i] = __builtin_amdgcn_wmma_f32_16x16x32_bf16(
          false, af, false, bfr, (short)0, acc[i], false, false);
    }
    __syncthreads();
  }

  const int col = colBase + wn * 16 + r16;
  const float bv = bias[col];
#pragma unroll
  for (int i = 0; i < 4; ++i) {
    const int row0 = rowBase + wm * 64 + i * 16 + h * 8;
#pragma unroll
    for (int p = 0; p < 8; ++p) {
      float v = acc[i][p] + bv;
      if (do_relu) v = fmaxf(v, 0.0f);
      C[(size_t)(row0 + p) * N + col] = v;
    }
  }
#endif
}

// ---------------------------------------------------------------------------
// In-place per-token top-k mask: one block/token, 256 threads, H = 32768.
// Post-ReLU values >= 0 -> uint bit order == float order; binary-search the
// k-th largest bit pattern with the row pinned in 128 VGPRs/thread.
// ---------------------------------------------------------------------------
__global__ __launch_bounds__(256) void topk_mask_kernel(
    float* __restrict__ feat, int H, const int* __restrict__ kptr)
{
  __shared__ unsigned cnt;
  const int n = blockIdx.x;
  const int t = threadIdx.x;
  const unsigned k = (unsigned)(*kptr);
  float* r = feat + (size_t)n * H;

  v4f rv[32];
  const v4f* rp = (const v4f*)(r + (size_t)t * 128);
#pragma unroll
  for (int j = 0; j < 32; ++j) rv[j] = rp[j];

  unsigned lo = 0u, hi = 0x7F800000u;
  while (hi - lo > 1u) {
    const unsigned mid = (lo + hi) >> 1;
    const float tv = __uint_as_float(mid);
    if (t == 0) cnt = 0u;
    __syncthreads();
    unsigned local = 0u;
#pragma unroll
    for (int j = 0; j < 32; ++j) {
      local += (rv[j].x >= tv) ? 1u : 0u;
      local += (rv[j].y >= tv) ? 1u : 0u;
      local += (rv[j].z >= tv) ? 1u : 0u;
      local += (rv[j].w >= tv) ? 1u : 0u;
    }
    atomicAdd(&cnt, local);
    __syncthreads();
    const unsigned c = cnt;
    __syncthreads();
    if (c >= k) lo = mid; else hi = mid;
  }
  const float T = __uint_as_float(lo);

  v4f* wp = (v4f*)(r + (size_t)t * 128);
#pragma unroll
  for (int j = 0; j < 32; ++j) {
    v4f v = rv[j], o;
    o.x = (v.x >= T) ? v.x : 0.0f;
    o.y = (v.y >= T) ? v.y : 0.0f;
    o.z = (v.z >= T) ? v.z : 0.0f;
    o.w = (v.w >= T) ? v.w : 0.0f;
    wp[j] = o;
  }
}

extern "C" void kernel_launch(void* const* d_in, const int* in_sizes, int n_in,
                              void* d_out, int out_size, void* d_ws, size_t ws_size,
                              hipStream_t stream) {
  (void)n_in; (void)out_size; (void)d_ws; (void)ws_size;

  const float* x     = (const float*)d_in[0];
  const float* W_enc = (const float*)d_in[1];
  const float* b_enc = (const float*)d_in[2];
  const float* W_dec = (const float*)d_in[3];
  const float* b_dec = (const float*)d_in[4];
  const int*   kptr  = (const int*)d_in[5];

  const int d_hid    = in_sizes[2];
  const int d_in_dim = in_sizes[4];
  const int n_tok    = in_sizes[0] / d_in_dim;

  float* recon = (float*)d_out;
  float* feat  = (float*)d_out + (size_t)n_tok * d_in_dim;

  dim3 g1(n_tok / BM, d_hid / BN);
  gemm_wmma<<<g1, dim3(256), 0, stream>>>(x, W_enc, b_enc, feat,
                                          n_tok, d_hid, d_in_dim, 1);

  topk_mask_kernel<<<dim3(n_tok), dim3(256), 0, stream>>>(feat, d_hid, kptr);

  dim3 g3(n_tok / BM, d_in_dim / BN);
  gemm_wmma<<<g3, dim3(256), 0, stream>>>(feat, W_dec, b_dec, recon,
                                          n_tok, d_in_dim, d_hid, 0);
}
